// IndependentKP_70042326663877
// MI455X (gfx1250) — compile-verified, matrix operand order checked by reading, not run on hardware
//
#include <hip/hip_runtime.h>
#include <hip/hip_bf16.h>
#include <stdint.h>

typedef __attribute__((ext_vector_type(16))) _Float16 v16h;
typedef __attribute__((ext_vector_type(8)))  float    v8f;

struct U8 { unsigned int u[8]; };

// ---------------------------------------------------------------------------
// Prep: gray = sum(RGB); pixel-unshuffle P=2 -> cs[b][c][y][x], c = px*2+py.
// ---------------------------------------------------------------------------
__global__ __launch_bounds__(256)
void prep_cs_kernel(const float* __restrict__ mosaic,
                    float* __restrict__ cs32, _Float16* __restrict__ cs16,
                    int nbatch) {
    int idx = blockIdx.x * blockDim.x + threadIdx.x;
    int total = nbatch * 4 * 256 * 256;
    if (idx >= total) return;
    int x = idx & 255;
    int y = (idx >> 8) & 255;
    int c = (idx >> 16) & 3;
    int b = idx >> 18;
    int py = c & 1, px = c >> 1;
    int Y = y * 2 + py, X = x * 2 + px;
    size_t base = ((size_t)b * 3) * (512 * 512) + (size_t)Y * 512 + X;
    float g = mosaic[base] + mosaic[base + 512 * 512] + mosaic[base + 2 * 512 * 512];
    cs32[idx] = g;
    cs16[idx] = (_Float16)g;
}

// ---------------------------------------------------------------------------
// Weight prep: f32 -> f16, with optional zero-padded row stride so every
// 32-k chunk of a row is 64 contiguous, 16B-aligned bytes (async-B128 ready).
// ---------------------------------------------------------------------------
__global__ __launch_bounds__(256)
void cast_pad_f16_kernel(const float* __restrict__ src, _Float16* __restrict__ dst,
                         int nrows, int kin, int kstride) {
    int i = blockIdx.x * blockDim.x + threadIdx.x;
    if (i >= nrows * kstride) return;
    int row = i / kstride;
    int k   = i - row * kstride;
    dst[i] = (k < kin) ? (_Float16)src[(size_t)row * kin + k] : (_Float16)0.0f;
}

// ---------------------------------------------------------------------------
// Implicit-GEMM conv via V_WMMA_F32_16X16X32_F16.
//   M tile = 128 output pixels per WG (16 per wave), N = NOUT (NT x 16), K
//   padded to 32. A: im2col gather (hoisted address math, coalesced u16
//   loads). B: f16 weights staged with global_load_async_to_lds_b128 (GVS),
//   synced via s_wait_asynccnt; masked fallback for partial tail chunks.
// X: f16 activations channel-major [chan][IH*IW]. Wh: f16 [s][NOUT][WSTR].
// ---------------------------------------------------------------------------
template <int CIN, int KSZ, int NOUT, int NT, int WSTR, bool RELU, bool HAS_S>
__global__ __launch_bounds__(256)
void conv_wmma_kernel(const _Float16* __restrict__ X,
                      const _Float16* __restrict__ Wh,
                      const float* __restrict__ Bf,
                      _Float16* __restrict__ Y,
                      int IH, int IW, int OH, int OW, int nbatch) {
    constexpr int KTOT   = CIN * KSZ * KSZ;
    constexpr int KSTEPS = (KTOT + 31) / 32;
    constexpr int ASTR   = 40;   // halves per LDS row (80 B: 16B-aligned, bank-friendly)
    constexpr int BSTR   = 40;

    __shared__ __align__(16) _Float16 As[128 * ASTR];
    __shared__ __align__(16) _Float16 Bs[64 * BSTR];

    const int tid  = threadIdx.x;
    const int lane = tid & 31;
    const int wv   = tid >> 5;

    const int sb = blockIdx.y;             // s*nbatch + b
    const int s  = sb / nbatch;
    const int b  = sb % nbatch;
    const int chan_base = HAS_S ? (sb * CIN) : (b * CIN);

    const int Mtot  = OH * OW;
    const int mbase = blockIdx.x * 128;
    const size_t ihw = (size_t)IH * IW;

    // ---- A-loader invariants: thread owns one pixel, 16 k-slots per step ----
    const int ml    = tid & 127;           // LDS row / local pixel
    const int kpar  = tid >> 7;            // k parity group (0/1)
    const int m     = mbase + ml;
    const bool mvalid = (m < Mtot);
    const int my = m / OW, mx = m - (m / OW) * OW;
    const _Float16* Xbase = X + (size_t)chan_base * ihw;

    // ---- async-B invariants: one B128 chunk per thread per k-step ----
    const int bn     = tid >> 2;                       // B row 0..63
    const int bnn    = (bn < NOUT) ? bn : (NOUT - 1);  // clamp (stay in bounds)
    const int bchunk = (tid & 3) << 4;                 // byte chunk within 64B
    const _Float16* wbase = Wh + (size_t)s * NOUT * WSTR;
    const unsigned long long wbase64 = (unsigned long long)(uintptr_t)wbase;
    const unsigned ldsB = (unsigned)(uintptr_t)&Bs[bn * BSTR] + (unsigned)bchunk;

    v8f acc[NT];
#pragma unroll
    for (int nt = 0; nt < NT; ++nt) acc[nt] = v8f{};

    const unsigned int* AsU = (const unsigned int*)As;
    const unsigned int* BsU = (const unsigned int*)Bs;
    const int koff_u = (lane >> 4) << 2;
    const int mrow   = wv * 16 + (lane & 15);

    for (int ks = 0; ks < KSTEPS; ++ks) {
        // ---- B tile: async DMA straight into LDS when the chunk is full ----
        const bool useAsync = ((ks * 32 + 32) * 2 <= WSTR * 2) && ((ks * 32 + 32) <= WSTR);
        if (useAsync) {
            unsigned goff = (unsigned)((bnn * WSTR + ks * 32) * 2 + bchunk);
            asm volatile("global_load_async_to_lds_b128 %0, %1, %2"
                         :: "v"(ldsB), "v"(goff), "s"(wbase64)
                         : "memory");
        } else {
            for (int i = tid; i < 64 * 32; i += 256) {
                int n  = i >> 5;
                int kl = i & 31;
                int kk = kl + ks * 32;
                _Float16 v = (_Float16)0.0f;
                if (n < NOUT && kk < KTOT) v = wbase[(size_t)n * WSTR + kk];
                Bs[n * BSTR + kl] = v;
            }
        }

        // ---- A tile: incremental im2col, coalesced u16 loads ----
        {
            int kk0 = ks * 32 + kpar;
            int cin, r;
            if constexpr (KSZ == 1) { cin = kk0; r = 0; }
            else { cin = kk0 / 9; r = kk0 - cin * 9; }
#pragma unroll
            for (int j = 0; j < 16; ++j) {
                int klocal = kpar + 2 * j;
                int kk = ks * 32 + klocal;
                _Float16 v = (_Float16)0.0f;
                if (mvalid && kk < KTOT) {
                    int dy, dx;
                    if constexpr (KSZ == 1) { dy = 0; dx = 0; }
                    else { dy = r / 3; dx = r - dy * 3; }
                    v = Xbase[(size_t)cin * ihw + (size_t)(my + dy) * IW + (mx + dx)];
                }
                As[ml * ASTR + klocal] = v;
                if constexpr (KSZ == 1) { cin += 2; }
                else { r += 2; if (r >= 9) { r -= 9; ++cin; } }
            }
        }

        if (useAsync) asm volatile("s_wait_asynccnt 0x0" ::: "memory");
        __syncthreads();

        // ---- A fragment (ISA 16-bit A 16x32 layout; pairs are contiguous) ----
        U8 au;
#pragma unroll
        for (int g = 0; g < 8; ++g)
            au.u[g] = AsU[mrow * (ASTR / 2) + (g + (g & 4) + koff_u)];
        v16h av = __builtin_bit_cast(v16h, au);

        // ---- NT WMMAs ----
#pragma unroll
        for (int nt = 0; nt < NT; ++nt) {
            U8 bu;
            const int nrow = nt * 16 + (lane & 15);
#pragma unroll
            for (int g = 0; g < 8; ++g)
                bu.u[g] = BsU[nrow * (BSTR / 2) + (g + (g & 4) + koff_u)];
            v16h bv = __builtin_bit_cast(v16h, bu);
            acc[nt] = __builtin_amdgcn_wmma_f32_16x16x32_f16(
                false, av, false, bv, (short)0, acc[nt], false, false);
        }
        __syncthreads();
    }

    // ---- bias + ReLU + store (C: VGPR r -> M=r+8*(lane>=16), N=lane&15) ----
#pragma unroll
    for (int nt = 0; nt < NT; ++nt) {
        int n = nt * 16 + (lane & 15);
        float bias = (n < NOUT) ? Bf[(size_t)s * NOUT + n] : 0.0f;
#pragma unroll
        for (int r = 0; r < 8; ++r) {
            int mm = mbase + wv * 16 + r + ((lane >> 4) << 3);
            float v = acc[nt][r] + bias;
            if (RELU) v = fmaxf(v, 0.0f);
            if (mm < Mtot && n < NOUT)
                Y[((size_t)sb * NOUT + n) * Mtot + mm] = (_Float16)v;
        }
    }
}

// ---------------------------------------------------------------------------
// Finalize: chans[s,b,y,x] = sum_{c,k} kern[s,b,c*9+k,y,x]*cs[b,c,1+y+dy,1+x+dx]
// pixel-shuffled into out[b, o, y*2+py, x*2+px], s = o*4 + py*2 + px; ch2 = 0.
// ---------------------------------------------------------------------------
__global__ __launch_bounds__(256)
void finalize_kernel(const _Float16* __restrict__ kern,
                     const float* __restrict__ cs32,
                     float* __restrict__ out, int nbatch) {
    int idx = blockIdx.x * blockDim.x + threadIdx.x;
    int total = nbatch * 252 * 252;
    if (idx >= total) return;
    int x = idx % 252;
    int t = idx / 252;
    int y = t % 252;
    int b = t / 252;

    float tile[4][9];
#pragma unroll
    for (int c = 0; c < 4; ++c)
#pragma unroll
        for (int dy = 0; dy < 3; ++dy)
#pragma unroll
            for (int dx = 0; dx < 3; ++dx)
                tile[c][dy * 3 + dx] =
                    cs32[(((size_t)b * 4 + c) * 256 + (1 + y + dy)) * 256 + (1 + x + dx)];

    const size_t mstride = 63504;  // 252*252
#pragma unroll
    for (int o = 0; o < 2; ++o) {
#pragma unroll
        for (int py = 0; py < 2; ++py) {
#pragma unroll
            for (int px = 0; px < 2; ++px) {
                int s = o * 4 + py * 2 + px;
                const _Float16* kp =
                    kern + ((size_t)(s * nbatch + b) * 36) * mstride + (size_t)y * 252 + x;
                float a = 0.0f;
#pragma unroll
                for (int c = 0; c < 4; ++c)
#pragma unroll
                    for (int k = 0; k < 9; ++k)
                        a += (float)kp[(size_t)(c * 9 + k) * mstride] * tile[c][k];
                out[(((size_t)b * 3 + o) * 504 + (y * 2 + py)) * 504 + (x * 2 + px)] = a;
            }
        }
    }
#pragma unroll
    for (int py = 0; py < 2; ++py)
#pragma unroll
        for (int px = 0; px < 2; ++px)
            out[(((size_t)b * 3 + 2) * 504 + (y * 2 + py)) * 504 + (x * 2 + px)] = 0.0f;
}

// ---------------------------------------------------------------------------
extern "C" void kernel_launch(void* const* d_in, const int* in_sizes, int n_in,
                              void* d_out, int out_size, void* d_ws, size_t ws_size,
                              hipStream_t stream) {
    const float* mosaic = (const float*)d_in[0];
    // d_in[1] (mask) unused by the reference.
    const float* w1 = (const float*)d_in[2];
    const float* b1 = (const float*)d_in[3];
    const float* w2 = (const float*)d_in[4];
    const float* b2 = (const float*)d_in[5];
    const float* w3 = (const float*)d_in[6];
    const float* b3 = (const float*)d_in[7];
    float* out = (float*)d_out;

    const int B = 4;
    const int UNK = 8;

    char* ws = (char*)d_ws;
    size_t off = 0;
    _Float16* cs16 = (_Float16*)(ws + off); off += (size_t)B * 4 * 65536 * sizeof(_Float16);
    float*    cs32 = (float*)(ws + off);    off += (size_t)B * 4 * 65536 * sizeof(float);
    _Float16* w1h  = (_Float16*)(ws + off); off += (size_t)UNK * 64 * 40 * sizeof(_Float16);   // padded rows
    _Float16* w2h  = (_Float16*)(ws + off); off += (size_t)UNK * 64 * 576 * sizeof(_Float16);
    _Float16* w3h  = (_Float16*)(ws + off); off += (size_t)UNK * 36 * 64 * sizeof(_Float16);
    off = (off + 255) & ~(size_t)255;
    _Float16* y1   = (_Float16*)(ws + off); off += (size_t)UNK * B * 64 * 64516 * sizeof(_Float16);
    _Float16* y2   = (_Float16*)(ws + off); off += (size_t)UNK * B * 64 * 63504 * sizeof(_Float16);
    _Float16* kern = (_Float16*)(ws + off); off += (size_t)UNK * B * 36 * 63504 * sizeof(_Float16);
    (void)off; (void)ws_size; (void)in_sizes; (void)n_in; (void)out_size;

    // 0) weights -> f16 (w1 rows padded 36->40 halves so chunks are 16B-aligned)
    cast_pad_f16_kernel<<<(UNK * 64 * 40  + 255) / 256, 256, 0, stream>>>(w1, w1h, UNK * 64, 36, 40);
    cast_pad_f16_kernel<<<(UNK * 64 * 576 + 255) / 256, 256, 0, stream>>>(w2, w2h, UNK * 64, 576, 576);
    cast_pad_f16_kernel<<<(UNK * 36 * 64  + 255) / 256, 256, 0, stream>>>(w3, w3h, UNK * 36, 64, 64);

    // 1) gray + pixel-unshuffle
    {
        int total = B * 4 * 65536;
        prep_cs_kernel<<<(total + 255) / 256, 256, 0, stream>>>(mosaic, cs32, cs16, B);
    }
    // 2) conv1: 4 -> 64 ch, 3x3 VALID, 256 -> 254   (K=36, 2 k-steps)
    {
        int mt = (64516 + 127) / 128;
        conv_wmma_kernel<4, 3, 64, 4, 40, true, false>
            <<<dim3(mt, UNK * B), 256, 0, stream>>>(cs16, w1h, b1, y1, 256, 256, 254, 254, B);
    }
    // 3) conv2: 64 -> 64 ch, 3x3 VALID, 254 -> 252  (K=576, 18 k-steps; dominant)
    {
        int mt = (63504 + 127) / 128;
        conv_wmma_kernel<64, 3, 64, 4, 576, true, true>
            <<<dim3(mt, UNK * B), 256, 0, stream>>>(y1, w2h, b2, y2, 254, 254, 252, 252, B);
    }
    // 4) 1x1 projection: 64 -> 36 ch (N padded to 48), bias, no ReLU
    {
        int mt = (63504 + 127) / 128;
        conv_wmma_kernel<64, 1, 36, 3, 64, false, true>
            <<<dim3(mt, UNK * B), 256, 0, stream>>>(y2, w3h, b3, kern, 252, 252, 252, 252, B);
    }
    // 5) 36-term contraction + pixel-shuffle + zero third channel
    {
        int total = B * 252 * 252;
        finalize_kernel<<<(total + 255) / 256, 256, 0, stream>>>(kern, cs32, out, B);
    }
}
